// RegionProposalNet_77446850282135
// MI455X (gfx1250) — matrix operand order repaired and checked by reading, not compiled, not run on hardware
//
#include <hip/hip_runtime.h>
#include <hip/hip_bf16.h>
#include <math.h>

// ---------------------------------------------------------------------------
// RPN for MI455X (gfx1250): bf16 WMMA implicit-GEMM conv3x3 (async-LDS double
// buffered) + fused WMMA heads, elementwise decode, per-image greedy NMS.
//   B=4, CIN=MID=512, FH=FW=64, NA=9, PRE=6000, POST=300
// ---------------------------------------------------------------------------

typedef __bf16 bf16;
typedef __bf16 bf16x16 __attribute__((ext_vector_type(16)));
typedef float  floatx8 __attribute__((ext_vector_type(8)));
typedef float  floatx4 __attribute__((ext_vector_type(4)));

union Frag32B { bf16x16 v; floatx4 q[2]; };

#define NBATCH   4
#define CIN      512
#define MID      512
#define FDIM     64
#define NPOS     4096              // 64*64
#define NANCH    36864             // 4096*9
#define PRE_N    6000
#define POST_N   300
#define NEG_INF  (-__builtin_inff())

// CDNA5 async global->LDS copy: each lane moves 16B; tracked by ASYNCcnt.
__device__ __forceinline__ void async_b128(unsigned int lds_off, const void* gaddr) {
  asm volatile("global_load_async_to_lds_b128 %0, %1, off"
               :: "v"(lds_off), "v"(gaddr)
               : "memory");
}
__device__ __forceinline__ void wait_async0() {
  asm volatile("s_wait_asynccnt 0x0" ::: "memory");
}

// ---------------------------------------------------------------------------
// Pack x [B,512,64,64] f32 (NCHW) -> xT [B,66,66,512] bf16, zero-padded halo.
// ---------------------------------------------------------------------------
__global__ void pack_x_kernel(const float* __restrict__ x, bf16* __restrict__ xT) {
  int idx = blockIdx.x * blockDim.x + threadIdx.x;
  const int total = NBATCH * 66 * 66 * CIN;
  if (idx >= total) return;
  int c   = idx & (CIN - 1);
  int rst = idx >> 9;
  int xx  = rst % 66; rst /= 66;
  int yy  = rst % 66;
  int bb  = rst / 66;
  float v = 0.f;
  if (yy >= 1 && yy <= FDIM && xx >= 1 && xx <= FDIM)
    v = x[(((size_t)bb * CIN + c) * FDIM + (yy - 1)) * FDIM + (xx - 1)];
  xT[idx] = (bf16)v;
}

// conv_w [co,ci,3,3] f32 -> Wt [tap][co][ci] bf16 (K-contiguous per column)
__global__ void pack_w_kernel(const float* __restrict__ w, bf16* __restrict__ Wt) {
  int idx = blockIdx.x * blockDim.x + threadIdx.x;
  const int total = 9 * MID * CIN;
  if (idx >= total) return;
  int ci = idx & (CIN - 1);
  int co = (idx >> 9) & (MID - 1);
  int t  = idx >> 18;
  Wt[idx] = (bf16)w[((size_t)co * CIN + ci) * 9 + t];
}

// Fuse cls_w[18,512] + reg_w[36,512] -> Wh [64][512] bf16 (rows 54..63 zero),
// and biases -> hb[64].
__global__ void pack_headw_kernel(const float* __restrict__ cls_w,
                                  const float* __restrict__ reg_w,
                                  const float* __restrict__ cls_b,
                                  const float* __restrict__ reg_b,
                                  bf16* __restrict__ Wh, float* __restrict__ hb) {
  int idx = blockIdx.x * blockDim.x + threadIdx.x;
  if (idx >= 64 * CIN) return;
  int ci = idx & (CIN - 1);
  int n  = idx >> 9;
  float v = 0.f;
  if (n < 18)       v = cls_w[(size_t)n * CIN + ci];
  else if (n < 54)  v = reg_w[(size_t)(n - 18) * CIN + ci];
  Wh[idx] = (bf16)v;
  if (idx < 64) {
    float bv = 0.f;
    if (idx < 18)      bv = cls_b[idx];
    else if (idx < 54) bv = reg_b[idx - 18];
    hb[idx] = bv;
  }
}

// ---------------------------------------------------------------------------
// 3x3 conv as implicit GEMM: M=(b,y,x)=16384, N=co=512, K=(tap,ci)=4608.
// Block (256 thr = 8 waves) -> 64M x 128N; wave -> 32M x 32N (2x2 WMMA frags).
// Per K-step (32), the unique 12KB tile (A: 64 rows x 64B, B: 128 cols x 64B)
// is staged into LDS with GLOBAL_LOAD_ASYNC_TO_LDS_B128, double buffered;
// waves read fragments with ds_load_b128 (80B pitch -> spread banks).
//   A frag: lane<16 row m=M0+lr bytes[0..16)|[32..48); lane>=16 [16..32)|[48..64)
//   B frag: lane<16 col n=N0+lr bytes[0..16)|[32..48); lane>=16 [16..32)|[48..64)
// ---------------------------------------------------------------------------
#define A_PITCH   80
#define A_BYTES   (64 * A_PITCH)            // 5120
#define B_BYTES   (128 * A_PITCH)           // 10240
#define BUF_BYTES (A_BYTES + B_BYTES)       // 15360

__global__ __launch_bounds__(256)
void conv3x3_wmma_kernel(const bf16* __restrict__ xT, const bf16* __restrict__ Wt,
                         const float* __restrict__ conv_b, bf16* __restrict__ h) {
  __shared__ __align__(16) char smem[2 * BUF_BYTES];

  const int mb    = blockIdx.x;           // 0..255 -> (b, y); M-tile = one x-row
  const int bIdx  = mb >> 6;
  const int y     = mb & 63;
  const int Nbase = blockIdx.y * 128;
  const int tid   = threadIdx.x;
  const int wv    = tid >> 5;
  const int wm    = wv & 1;               // 0..1  (M: 32 each)
  const int wn    = wv >> 1;              // 0..3  (N: 32 each)
  const int lane  = tid & 31;
  const int lr    = lane & 15;
  const int lh    = lane >> 4;

  // --- staging assignment: thread -> 1 A-chunk + 2 B-chunks (16B each) ---
  const int cRow = tid >> 2;              // 0..63 (A row / B col block 0)
  const int cCh  = tid & 3;               // 16B chunk within 64B of K data
  const unsigned int smem0 = (unsigned int)(uintptr_t)smem;   // LDS byte offset
  const unsigned int aLds  = smem0 + cRow * A_PITCH + cCh * 16;
  const unsigned int bLds  = smem0 + A_BYTES + cRow * A_PITCH + cCh * 16;

  // --- consumption offsets (within a buffer) ---
  const int rowA = wm * 32 + lr;
  const int colB = wn * 32 + lr;
  const unsigned int aOff = rowA * A_PITCH + lh * 16;            // +1280 fm, +32 ch1
  const unsigned int bOff = A_BYTES + colB * A_PITCH + lh * 16;  // +1280 fn, +32 ch1

  floatx8 acc[2][2] = {};

  auto stage = [&](int it2, int buf) {
    const int t2  = it2 >> 4;             // tap 0..8
    const int kb2 = (it2 & 15) << 5;      // ci base 0..480
    const int dy2 = t2 / 3, dx2 = t2 - dy2 * 3;
    const unsigned int bb = (unsigned int)(buf * BUF_BYTES);
    const bf16* gA = xT + ((size_t)((bIdx * 66 + y + dy2) * 66) + (cRow + dx2)) * CIN
                        + kb2 + cCh * 8;
    const bf16* gB = Wt + ((size_t)(t2 * MID + Nbase + cRow)) * CIN + kb2 + cCh * 8;
    async_b128(aLds + bb, gA);                     // A rows 0..63
    async_b128(bLds + bb, gB);                     // B cols 0..63
    async_b128(bLds + bb + 64 * A_PITCH, gB + (size_t)64 * CIN);  // B cols 64..127
  };

  stage(0, 0);
  wait_async0();
  __syncthreads();

  for (int it = 0; it < 144; ++it) {
    const int buf = it & 1;
    if (it + 1 < 144) stage(it + 1, buf ^ 1);

    const char* Sb = smem + buf * BUF_BYTES;
    Frag32B A0, A1, B0, B1;
    A0.q[0] = *(const floatx4*)(Sb + aOff);
    A0.q[1] = *(const floatx4*)(Sb + aOff + 32);
    A1.q[0] = *(const floatx4*)(Sb + aOff + 1280);
    A1.q[1] = *(const floatx4*)(Sb + aOff + 1280 + 32);
    B0.q[0] = *(const floatx4*)(Sb + bOff);
    B0.q[1] = *(const floatx4*)(Sb + bOff + 32);
    B1.q[0] = *(const floatx4*)(Sb + bOff + 1280);
    B1.q[1] = *(const floatx4*)(Sb + bOff + 1280 + 32);

    acc[0][0] = __builtin_amdgcn_wmma_f32_16x16x32_bf16(false, A0.v, false, B0.v,
                                                        (short)0, acc[0][0], false, false);
    acc[0][1] = __builtin_amdgcn_wmma_f32_16x16x32_bf16(false, A0.v, false, B1.v,
                                                        (short)0, acc[0][1], false, false);
    acc[1][0] = __builtin_amdgcn_wmma_f32_16x16x32_bf16(false, A1.v, false, B0.v,
                                                        (short)0, acc[1][0], false, false);
    acc[1][1] = __builtin_amdgcn_wmma_f32_16x16x32_bf16(false, A1.v, false, B1.v,
                                                        (short)0, acc[1][1], false, false);

    wait_async0();          // next buffer landed (wave-local) ...
    __syncthreads();        // ... block-wide
  }

  // C layout: VGPR r, lane L: m = r + 8*(L>=16), n = L%16
  for (int fm = 0; fm < 2; ++fm)
    for (int fn = 0; fn < 2; ++fn) {
      const int   n    = Nbase + wn * 32 + fn * 16 + lr;
      const float bias = conv_b[n];
      const int   xb   = wm * 32 + fm * 16 + lh * 8;
#pragma unroll
      for (int r = 0; r < 8; ++r) {
        float v = acc[fm][fn][r] + bias;
        v = v > 0.f ? v : 0.f;                      // ReLU
        const int xx = xb + r;
        h[((size_t)(bIdx * NPOS + y * FDIM + xx)) * MID + n] = (bf16)v;
      }
    }
}

// ---------------------------------------------------------------------------
// Fused heads (1x1 convs) as GEMM: M=16384, N=64 (54 live), K=512.
// Block -> 128M x 64N (8 waves: 4M x 2N of 32x32); direct global fragment
// loads (tiny GEMM, fully L2 resident).
// ---------------------------------------------------------------------------
__global__ __launch_bounds__(256)
void heads_wmma_kernel(const bf16* __restrict__ h, const bf16* __restrict__ Wh,
                       const float* __restrict__ hb, float* __restrict__ hout) {
  const int Mbase = blockIdx.x * 128;
  const int wv    = threadIdx.x >> 5;
  const int wm    = wv >> 1;              // 0..3
  const int wn    = wv & 1;               // 0..1
  const int lane  = threadIdx.x & 31;
  const int lr    = lane & 15;
  const int lh8   = (lane >> 4) * 8;
  const int lh16  = (lane >> 4) * 16;

  floatx8 acc[2][2] = {};
  const bf16* arow = h  + ((size_t)(Mbase + wm * 32 + lr)) * MID + lh8;
  const bf16* brow = Wh + ((size_t)(wn * 32 + lr)) * MID + lh8;

  for (int kb = 0; kb < MID; kb += 32) {
    Frag32B A0, A1, B0, B1;
    A0.q[0] = *(const floatx4*)(arow + kb);
    A0.q[1] = *(const floatx4*)(arow + kb + 16);
    A1.q[0] = *(const floatx4*)(arow + 16 * MID + kb);
    A1.q[1] = *(const floatx4*)(arow + 16 * MID + kb + 16);
    B0.q[0] = *(const floatx4*)(brow + kb);
    B0.q[1] = *(const floatx4*)(brow + kb + 16);
    B1.q[0] = *(const floatx4*)(brow + 16 * MID + kb);
    B1.q[1] = *(const floatx4*)(brow + 16 * MID + kb + 16);
    acc[0][0] = __builtin_amdgcn_wmma_f32_16x16x32_bf16(false, A0.v, false, B0.v,
                                                        (short)0, acc[0][0], false, false);
    acc[0][1] = __builtin_amdgcn_wmma_f32_16x16x32_bf16(false, A0.v, false, B1.v,
                                                        (short)0, acc[0][1], false, false);
    acc[1][0] = __builtin_amdgcn_wmma_f32_16x16x32_bf16(false, A1.v, false, B0.v,
                                                        (short)0, acc[1][0], false, false);
    acc[1][1] = __builtin_amdgcn_wmma_f32_16x16x32_bf16(false, A1.v, false, B1.v,
                                                        (short)0, acc[1][1], false, false);
  }

  for (int fm = 0; fm < 2; ++fm)
    for (int fn = 0; fn < 2; ++fn) {
      const int   n    = wn * 32 + fn * 16 + lr;
      const float bias = hb[n];
#pragma unroll
      for (int r = 0; r < 8; ++r) {
        const int m = Mbase + wm * 32 + fm * 16 + lh8 + r;
        hout[(size_t)m * 64 + n] = acc[fm][fn][r] + bias;
      }
    }
}

// ---------------------------------------------------------------------------
// Per-anchor: emit out_classifier/out_offsets, anchors, softmax score,
// decode + clip + min-size filter -> boxes/scores for NMS.
// ---------------------------------------------------------------------------
__global__ void postprocess_kernel(const float* __restrict__ hout,
                                   const int* __restrict__ imh_p,
                                   const int* __restrict__ imw_p,
                                   float* __restrict__ outO, float* __restrict__ outC,
                                   float* __restrict__ anc,
                                   float* __restrict__ boxes, float* __restrict__ scores) {
  int tid = blockIdx.x * blockDim.x + threadIdx.x;
  if (tid >= NBATCH * NANCH) return;
  int b   = tid / NANCH;
  int rem = tid - b * NANCH;
  int pos = rem / 9;
  int a   = rem - pos * 9;
  int m   = b * NPOS + pos;
  const float* hr = hout + (size_t)m * 64;

  float c0 = hr[a * 2 + 0], c1 = hr[a * 2 + 1];
  float o0 = hr[18 + a * 4 + 0], o1 = hr[18 + a * 4 + 1];
  float o2 = hr[18 + a * 4 + 2], o3 = hr[18 + a * 4 + 3];

  outC[(size_t)tid * 2 + 0] = c0;  outC[(size_t)tid * 2 + 1] = c1;
  outO[(size_t)tid * 4 + 0] = o0;  outO[(size_t)tid * 4 + 1] = o1;
  outO[(size_t)tid * 4 + 2] = o2;  outO[(size_t)tid * 4 + 3] = o3;

  // anchor (base 16, ratios {.5,1,2}, scales {8,16,32}), stride 16
  const float ratios[3] = {0.5f, 1.0f, 2.0f};
  const float scales[3] = {8.f, 16.f, 32.f};
  int ri = a / 3, si = a % 3;
  float sr = sqrtf(ratios[ri]);
  float ah_ = 16.f * scales[si] / sr;
  float aw_ = 16.f * scales[si] * sr;
  int   gy  = pos >> 6, gx = pos & 63;
  float sx = 16.f * (float)gx, sy = 16.f * (float)gy;
  float ax1 = 7.5f - 0.5f * (aw_ - 1.f) + sx;
  float ay1 = 7.5f - 0.5f * (ah_ - 1.f) + sy;
  float ax2 = 7.5f + 0.5f * (aw_ - 1.f) + sx;
  float ay2 = 7.5f + 0.5f * (ah_ - 1.f) + sy;
  if (b == 0) {
    anc[(size_t)rem * 4 + 0] = ax1;  anc[(size_t)rem * 4 + 1] = ay1;
    anc[(size_t)rem * 4 + 2] = ax2;  anc[(size_t)rem * 4 + 3] = ay2;
  }

  float s = 1.f / (1.f + expf(c0 - c1));   // softmax[...,1]

  // decode (legacy +1 convention)
  float aw  = fmaxf(ax2 - ax1 + 1.f, 1e-5f);
  float ah  = fmaxf(ay2 - ay1 + 1.f, 1e-5f);
  float acx = 0.5f * (ax1 + ax2), acy = 0.5f * (ay1 + ay2);
  float tw  = aw * expf(o2), th = ah * expf(o3);
  float tcx = acx + o0 * aw,  tcy = acy + o1 * ah;
  float imw = (float)imw_p[0], imh = (float)imh_p[0];
  float x1 = fminf(fmaxf(tcx - 0.5f * (tw - 1.f), 0.f), imw - 1.f);
  float y1 = fminf(fmaxf(tcy - 0.5f * (th - 1.f), 0.f), imh - 1.f);
  float x2 = fminf(fmaxf(tcx + 0.5f * (tw - 1.f), 0.f), imw - 1.f);
  float y2 = fminf(fmaxf(tcy + 0.5f * (th - 1.f), 0.f), imh - 1.f);

  bool valid = ((x2 - x1 + 1.f) >= 16.f) && ((y2 - y1 + 1.f) >= 16.f);
  boxes[(size_t)tid * 4 + 0] = x1;  boxes[(size_t)tid * 4 + 1] = y1;
  boxes[(size_t)tid * 4 + 2] = x2;  boxes[(size_t)tid * 4 + 3] = y2;
  scores[tid] = valid ? s : NEG_INF;
}

// ---------------------------------------------------------------------------
// Greedy NMS, one workgroup (1024 thr) per batch image.
// Bisection reproduces the top-PRE score cutoff; then 300 rounds of
// block-argmax + IoU suppression.  Pad = first selected box; rois_idx = b.
// ---------------------------------------------------------------------------
__global__ __launch_bounds__(1024)
void nms_kernel(const float* __restrict__ boxes_all, float* __restrict__ scores_all,
                float* __restrict__ rois, int* __restrict__ rois_idx) {
  const int b = blockIdx.x;
  const float* bx = boxes_all + (size_t)b * NANCH * 4;
  float* sc = scores_all + (size_t)b * NANCH;
  const int tid = threadIdx.x;

  __shared__ float sv[1024];
  __shared__ int   si[1024];
  __shared__ int   scnt;
  __shared__ float selb[5];
  __shared__ float padb[4];
  __shared__ int   selok;

  // ---- top-PRE threshold via bisection on score value in [0,1] ----
  float lo = 0.f, hi = 1.f;
  for (int it = 0; it < 32; ++it) {
    float mid = 0.5f * (lo + hi);
    if (tid == 0) scnt = 0;
    __syncthreads();
    int c = 0;
    for (int i = tid; i < NANCH; i += 1024)
      if (sc[i] > mid) ++c;
    atomicAdd(&scnt, c);
    __syncthreads();
    int total = scnt;
    __syncthreads();
    if (total > PRE_N) lo = mid; else hi = mid;
  }
  for (int i = tid; i < NANCH; i += 1024)
    if (!(sc[i] > lo)) sc[i] = NEG_INF;
  __syncthreads();

  // ---- 300 rounds of argmax + suppress ----
  for (int it = 0; it < POST_N; ++it) {
    float bv = NEG_INF;
    int   bi = 0;
    for (int i = tid; i < NANCH; i += 1024) {
      float v = sc[i];
      if (v > bv) { bv = v; bi = i; }
    }
    sv[tid] = bv; si[tid] = bi;
    __syncthreads();
    for (int off = 512; off > 0; off >>= 1) {
      if (tid < off) {
        if (sv[tid + off] > sv[tid] ||
            (sv[tid + off] == sv[tid] && si[tid + off] < si[tid])) {
          sv[tid] = sv[tid + off];
          si[tid] = si[tid + off];
        }
      }
      __syncthreads();
    }
    if (tid == 0) {
      int  i  = si[0];
      bool ok = sv[0] > NEG_INF;
      float x1 = bx[(size_t)i * 4 + 0], y1 = bx[(size_t)i * 4 + 1];
      float x2 = bx[(size_t)i * 4 + 2], y2 = bx[(size_t)i * 4 + 3];
      if (it == 0) { padb[0] = x1; padb[1] = y1; padb[2] = x2; padb[3] = y2; }
      float ox1 = ok ? x1 : padb[0], oy1 = ok ? y1 : padb[1];
      float ox2 = ok ? x2 : padb[2], oy2 = ok ? y2 : padb[3];
      size_t ro = ((size_t)b * POST_N + it) * 4;
      rois[ro + 0] = ox1; rois[ro + 1] = oy1; rois[ro + 2] = ox2; rois[ro + 3] = oy2;
      rois_idx[b * POST_N + it] = b;
      if (ok) sc[i] = NEG_INF;
      selb[0] = x1; selb[1] = y1; selb[2] = x2; selb[3] = y2;
      selb[4] = (x2 - x1) * (y2 - y1);
      selok = ok ? 1 : 0;
    }
    __syncthreads();
    if (selok) {
      float ax1 = selb[0], ay1 = selb[1], ax2 = selb[2], ay2 = selb[3], aar = selb[4];
      for (int i = tid; i < NANCH; i += 1024) {
        float v = sc[i];
        if (v == NEG_INF) continue;
        float x1 = bx[(size_t)i * 4 + 0], y1 = bx[(size_t)i * 4 + 1];
        float x2 = bx[(size_t)i * 4 + 2], y2 = bx[(size_t)i * 4 + 3];
        float xx1 = fmaxf(ax1, x1), yy1 = fmaxf(ay1, y1);
        float xx2 = fminf(ax2, x2), yy2 = fminf(ay2, y2);
        float inter = fmaxf(xx2 - xx1, 0.f) * fmaxf(yy2 - yy1, 0.f);
        float area  = (x2 - x1) * (y2 - y1);
        float iou   = inter / (aar + area - inter + 1e-12f);
        if (iou > 0.7f) sc[i] = NEG_INF;
      }
    }
    __syncthreads();
  }
}

// ---------------------------------------------------------------------------
extern "C" void kernel_launch(void* const* d_in, const int* in_sizes, int n_in,
                              void* d_out, int out_size, void* d_ws, size_t ws_size,
                              hipStream_t stream) {
  const float* x      = (const float*)d_in[0];
  const float* conv_w = (const float*)d_in[1];
  const float* conv_b = (const float*)d_in[2];
  const float* cls_w  = (const float*)d_in[3];
  const float* cls_b  = (const float*)d_in[4];
  const float* reg_w  = (const float*)d_in[5];
  const float* reg_b  = (const float*)d_in[6];
  const int*   im_h   = (const int*)d_in[7];
  const int*   im_w   = (const int*)d_in[8];

  // workspace carve (all rewritten every launch -> deterministic)
  char* wsp = (char*)d_ws;
  auto carve = [&](size_t bytes) {
    char* p = wsp;
    wsp += (bytes + 255) & ~(size_t)255;
    return p;
  };
  bf16*  xT     = (bf16*)carve((size_t)NBATCH * 66 * 66 * CIN * sizeof(bf16));
  bf16*  Wt     = (bf16*)carve((size_t)9 * MID * CIN * sizeof(bf16));
  bf16*  h      = (bf16*)carve((size_t)NBATCH * NPOS * MID * sizeof(bf16));
  bf16*  Wh     = (bf16*)carve((size_t)64 * CIN * sizeof(bf16));
  float* hb     = (float*)carve(64 * sizeof(float));
  float* hout   = (float*)carve((size_t)NBATCH * NPOS * 64 * sizeof(float));
  float* scores = (float*)carve((size_t)NBATCH * NANCH * sizeof(float));
  float* boxes  = (float*)carve((size_t)NBATCH * NANCH * 4 * sizeof(float));

  // outputs (concatenated flat, reference return order)
  float* outO = (float*)d_out;                       // [4,36864,4]
  float* outC = outO + (size_t)NBATCH * NANCH * 4;   // [4,36864,2]
  float* rois = outC + (size_t)NBATCH * NANCH * 2;   // [4,300,4]
  int*   ridx = (int*)(rois + (size_t)NBATCH * POST_N * 4);  // [4,300] int32
  float* anc  = (float*)(ridx + (size_t)NBATCH * POST_N);    // [36864,4]

  const int xt_n = NBATCH * 66 * 66 * CIN;
  pack_x_kernel<<<(xt_n + 255) / 256, 256, 0, stream>>>(x, xT);
  pack_w_kernel<<<(9 * MID * CIN + 255) / 256, 256, 0, stream>>>(conv_w, Wt);
  pack_headw_kernel<<<(64 * CIN + 255) / 256, 256, 0, stream>>>(cls_w, reg_w, cls_b,
                                                                reg_b, Wh, hb);
  conv3x3_wmma_kernel<<<dim3(NBATCH * FDIM, MID / 128), 256, 0, stream>>>(xT, Wt,
                                                                          conv_b, h);
  heads_wmma_kernel<<<(NBATCH * NPOS) / 128, 256, 0, stream>>>(h, Wh, hb, hout);
  postprocess_kernel<<<(NBATCH * NANCH + 255) / 256, 256, 0, stream>>>(
      hout, im_h, im_w, outO, outC, anc, boxes, scores);
  nms_kernel<<<NBATCH, 1024, 0, stream>>>(boxes, scores, rois, ridx);
}
